// TinyLlamaGQA_16587163697405
// MI455X (gfx1250) — matrix operand order repaired and checked by reading, not compiled, not run on hardware
//
#include <hip/hip_runtime.h>
#include <hip/hip_bf16.h>

typedef __attribute__((ext_vector_type(16))) __bf16 v16bf;
typedef __attribute__((ext_vector_type(8)))  float  v8f;

#define BSZ   4
#define SEQ   2048
#define HID   768
#define NH    12
#define NKV   4
#define NREP  3
#define HD    64
#define BS    (BSZ*SEQ)          // 8192 tokens
#define NCOL  (HID + 256 + 256)  // fused qkv output columns = 1280
#define NEGBIG (-1.0e30f)

static __device__ __forceinline__ v8f wmma_bf16(v16bf a, v16bf b, v8f c) {
  return __builtin_amdgcn_wmma_f32_16x16x32_bf16(false, a, false, b, (short)0, c,
                                                 false, false);
}

// ---- CDNA5 async LDS<->global helpers (inline asm; ASYNCcnt-tracked) -------
// LDS byte offset = low 32 bits of the generic pointer (ISA 10.2: flat->LDS
// uses addr[31:0] as the wave-relative LDS address).
static __device__ __forceinline__ uint32_t lds_off(const void* p) {
  return (uint32_t)(uintptr_t)p;
}
static __device__ __forceinline__ void async_load_b128(uint32_t dst_lds,
                                                       const void* src_gbl) {
  asm volatile("global_load_async_to_lds_b128 %0, %1, off"
               :: "v"(dst_lds), "v"(src_gbl) : "memory");
}
static __device__ __forceinline__ void async_store_b128(void* dst_gbl,
                                                        uint32_t src_lds) {
  asm volatile("global_store_async_from_lds_b128 %0, %1, off"
               :: "v"(dst_gbl), "v"(src_lds) : "memory");
}
static __device__ __forceinline__ void wait_async() {
  asm volatile("s_wait_asynccnt 0" ::: "memory");
}

// ---- fragment loaders per CDNA5 ISA layouts (wave32) -----------------------
// A (16x32 bf16): lane<16 holds row M=lane, K in {0..7} (elems 0-7) and
// {16..23} (elems 8-15); lane>=16 holds row M=lane-16, K in {8..15},{24..31}.
static __device__ __forceinline__ v16bf load_a_f32(const float* row, int kbase, bool hi) {
  v16bf a;
  const float* p = row + kbase + (hi ? 8 : 0);
#pragma unroll
  for (int e = 0; e < 8; ++e) a[e] = (__bf16)p[e];
#pragma unroll
  for (int e = 0; e < 8; ++e) a[e + 8] = (__bf16)p[16 + e];
  return a;
}
static __device__ __forceinline__ v16bf load_a_bf16(const __bf16* row, int kbase, bool hi) {
  v16bf a;
  const __bf16* p = row + kbase + (hi ? 8 : 0);
#pragma unroll
  for (int e = 0; e < 8; ++e) a[e] = p[e];
#pragma unroll
  for (int e = 0; e < 8; ++e) a[e + 8] = p[16 + e];
  return a;
}
// B (32x16 bf16): lane holds column N=lane&15; lane<16 K in {0..7},{16..23},
// lane>=16 K in {8..15},{24..31} (same K order as A).
static __device__ __forceinline__ v16bf load_b_f32(const float* wrow, int kbase, bool hi) {
  v16bf b;
  const float* p = wrow + kbase + (hi ? 8 : 0);
#pragma unroll
  for (int e = 0; e < 8; ++e) b[e] = (__bf16)p[e];
#pragma unroll
  for (int e = 0; e < 8; ++e) b[e + 8] = (__bf16)p[16 + e];
  return b;
}
static __device__ __forceinline__ v16bf load_b_bf16(const __bf16* wrow, int kbase, bool hi) {
  v16bf b;
  const __bf16* p = wrow + kbase + (hi ? 8 : 0);
#pragma unroll
  for (int e = 0; e < 8; ++e) b[e] = p[e];
#pragma unroll
  for (int e = 0; e < 8; ++e) b[e + 8] = p[16 + e];
  return b;
}

// ============================================================================
// Kernel 1: fused QKV projection (bf16 WMMA, f32 accum) + RoPE, scatter to
// bf16 buffers: q,k in (b, head, s, d); V stored TRANSPOSED (b, kvh, d, s) so
// the attention P@V B-fragment becomes a contiguous load.
// ============================================================================
__global__ __launch_bounds__(128) void qkv_rope_kernel(
    const float* __restrict__ X, const float* __restrict__ wq,
    const float* __restrict__ wk, const float* __restrict__ wv,
    const float* __restrict__ fcos, const float* __restrict__ fsin,
    __bf16* __restrict__ qb, __bf16* __restrict__ kb, __bf16* __restrict__ vbT) {
  const int wave = threadIdx.x >> 5;
  const int lane = threadIdx.x & 31;
  const bool hi  = lane >= 16;
  const int tile = blockIdx.x * 4 + wave;
  const int ct = tile % (NCOL / 16);
  const int rt = tile / (NCOL / 16);

  const int mloc = lane & 15;
  const float* xrow = X + (size_t)(rt * 16 + mloc) * HID;

  const int j = ct * 16 + mloc;  // fused output column
  const float* wrow;
  if (j < HID)            wrow = wq + (size_t)j * HID;
  else if (j < HID + 256) wrow = wk + (size_t)(j - HID) * HID;
  else                    wrow = wv + (size_t)(j - HID - 256) * HID;

  v8f c = {};
#pragma unroll 4
  for (int kk = 0; kk < HID; kk += 32) {
    __builtin_prefetch(xrow + kk + 128, 0, 3);   // global_prefetch_b8
    __builtin_prefetch(wrow + kk + 128, 0, 3);
    c = wmma_bf16(load_a_f32(xrow, kk, hi), load_b_f32(wrow, kk, hi), c);
  }

  // RoPE (q and k columns only) + scatter stores.
  const bool isqk = j < HID + 256;
  const int dd = j & 63;       // dim within head
  const int p  = dd >> 1;      // rotary pair index
  const bool odd = j & 1;
#pragma unroll
  for (int r = 0; r < 8; ++r) {
    const int t  = rt * 16 + r + (hi ? 8 : 0);  // global token
    const int ss = t & (SEQ - 1);
    const int bi = t >> 11;                     // t / SEQ
    float v  = c[r];
    float pv = __shfl_xor(v, 1, 32);            // rotary partner column
    float ov = v;
    if (isqk) {
      float cv = fcos[ss * 32 + p];
      float sv = fsin[ss * 32 + p];
      ov = odd ? (pv * sv + v * cv) : (v * cv - pv * sv);
    }
    if (j < HID) {
      qb[(((size_t)bi * NH + (j >> 6)) * SEQ + ss) * HD + dd] = (__bf16)ov;
    } else if (j < HID + 256) {
      kb[(((size_t)bi * NKV + ((j - HID) >> 6)) * SEQ + ss) * HD + dd] = (__bf16)ov;
    } else {
      // transposed: [b][kvh][d][s]
      vbT[(((size_t)bi * NKV + ((j - HID - 256) >> 6)) * HD + dd) * SEQ + ss] = (__bf16)ov;
    }
  }
}

// ============================================================================
// Kernel 2: causal GQA attention for one (b, h, 16-row q tile) per block.
// 4 waves stripe 32-key chunks. K chunks staged into LDS with
// global_load_async_to_lds_b128; attn tiles written with
// global_store_async_from_lds_b128 directly from the LDS transpose buffer.
// Pass 1: WMMA scores + online softmax stats. Pass 2: recompute, write attn,
// P@V via WMMA against the d-major V buffer.
// ============================================================================
__global__ __launch_bounds__(128) void attn_kernel(
    const __bf16* __restrict__ qb, const __bf16* __restrict__ kb,
    const __bf16* __restrict__ vbT, float* __restrict__ attn,
    __bf16* __restrict__ ctx) {
  __shared__ __align__(16) __bf16 s_kt[4][32 * HD];  // per-wave K chunk (4KB)
  __shared__ __align__(16) float s_pt[4][16 * 32];   // per-wave P tile
  __shared__ float s_stats[4][16][2];                // per-wave (m, l)
  __shared__ float s_m[16], s_invl[16];
  __shared__ float s_acc[16 * 64];                   // output accumulator

  const int tid  = threadIdx.x;
  const int wave = tid >> 5;
  const int lane = tid & 31;
  const bool hi  = lane >= 16;
  const int nl   = lane & 15;

  const int qt = blockIdx.x & 127;
  const int h  = (blockIdx.x >> 7) % NH;
  const int b  = blockIdx.x / (128 * NH);
  const int hkv = h / NREP;

  const __bf16* qbase = qb + ((size_t)b * NH + h) * SEQ * HD;
  const __bf16* kbase = kb + ((size_t)b * NKV + hkv) * SEQ * HD;
  const __bf16* vbase = vbT + ((size_t)b * NKV + hkv) * HD * SEQ;  // d-major
  float* attn_row0 = attn + ((size_t)(b * NH + h) * SEQ + (size_t)qt * 16) * SEQ;

  const uint32_t kt_lds = lds_off(&s_kt[wave][0]);
  const uint32_t pt_lds = lds_off(&s_pt[wave][0]);

  for (int i = tid; i < 16 * 64; i += 128) s_acc[i] = 0.0f;

  const int nvalid = (qt + 1) * 16;          // causal frontier
  const int nchunk = (nvalid + 31) >> 5;     // 32-key chunks
  const int kstart = nchunk * 32;            // zero-fill from here

  // zero the causally-dead attn columns (float4 stores)
  {
    const int n4 = (SEQ - kstart) >> 2;
    const float4 z4 = make_float4(0.f, 0.f, 0.f, 0.f);
    for (int i = tid; i < 16 * n4; i += 128) {
      const int r = i / n4, c4 = i % n4;
      ((float4*)(attn_row0 + (size_t)r * SEQ + kstart))[c4] = z4;
    }
  }

  // Q fragments (K = head dim 0..31, 32..63), reused for both passes.
  const __bf16* qrow = qbase + (size_t)(qt * 16 + nl) * HD;
  const v16bf aq0 = load_a_bf16(qrow, 0, hi);
  const v16bf aq1 = load_a_bf16(qrow, 32, hi);

  const float scale = 0.125f;  // 1/sqrt(64)

  // ---- pass 1: online softmax stats --------------------------------------
  float m[8], l[8];
#pragma unroll
  for (int r = 0; r < 8; ++r) { m[r] = NEGBIG; l[r] = 0.0f; }

  for (int kc = wave; kc < nchunk; kc += 4) {
    // stage K chunk (32 rows x 64 bf16 = 4KB) into LDS asynchronously
    const char* gk = (const char*)(kbase + (size_t)(kc * 32) * HD);
#pragma unroll
    for (int blk = 0; blk < 8; ++blk)
      async_load_b128(kt_lds + blk * 512 + lane * 16, gk + blk * 512 + lane * 16);
    wait_async();

#pragma unroll
    for (int half = 0; half < 2; ++half) {
      const int k0 = kc * 32 + half * 16;
      const __bf16* krow = &s_kt[wave][(half * 16 + nl) * HD];
      v8f sc = {};
      sc = wmma_bf16(aq0, load_b_bf16(krow, 0, hi), sc);
      sc = wmma_bf16(aq1, load_b_bf16(krow, 32, hi), sc);
#pragma unroll
      for (int r = 0; r < 8; ++r) {
        const int qi = qt * 16 + r + (hi ? 8 : 0);
        const int kj = k0 + nl;
        float sv = (kj > qi) ? NEGBIG : sc[r] * scale;
        float tmax = sv;
#pragma unroll
        for (int off = 1; off <= 8; off <<= 1)
          tmax = fmaxf(tmax, __shfl_xor(tmax, off, 32));
        const float mn = fmaxf(m[r], tmax);
        float pe = (kj > qi) ? 0.0f : __expf(sv - mn);
#pragma unroll
        for (int off = 1; off <= 8; off <<= 1) pe += __shfl_xor(pe, off, 32);
        l[r] = l[r] * __expf(m[r] - mn) + pe;
        m[r] = mn;
      }
    }
  }
  if (nl == 0) {
#pragma unroll
    for (int r = 0; r < 8; ++r) {
      s_stats[wave][r + (hi ? 8 : 0)][0] = m[r];
      s_stats[wave][r + (hi ? 8 : 0)][1] = l[r];
    }
  }
  __syncthreads();
  if (tid < 16) {
    float M = NEGBIG;
    for (int w = 0; w < 4; ++w) M = fmaxf(M, s_stats[w][tid][0]);
    float L = 0.0f;
    for (int w = 0; w < 4; ++w) L += s_stats[w][tid][1] * __expf(s_stats[w][tid][0] - M);
    s_m[tid] = M;
    s_invl[tid] = 1.0f / L;
  }
  __syncthreads();

  // ---- pass 2: write attn + P@V ------------------------------------------
  v8f oc0 = {}, oc1 = {}, oc2 = {}, oc3 = {};
  const int niter = (nchunk + 3) >> 2;   // uniform trip count for barriers
  for (int it = 0; it < niter; ++it) {
    const int kc = it * 4 + wave;
    const bool act = kc < nchunk;        // wave-uniform predicate
    if (act) {
      const char* gk = (const char*)(kbase + (size_t)(kc * 32) * HD);
#pragma unroll
      for (int blk = 0; blk < 8; ++blk)
        async_load_b128(kt_lds + blk * 512 + lane * 16, gk + blk * 512 + lane * 16);
      wait_async();  // K chunk in LDS; also fences prior attn async stores
#pragma unroll
      for (int half = 0; half < 2; ++half) {
        const int k0 = kc * 32 + half * 16;
        const __bf16* krow = &s_kt[wave][(half * 16 + nl) * HD];
        v8f sc = {};
        sc = wmma_bf16(aq0, load_b_bf16(krow, 0, hi), sc);
        sc = wmma_bf16(aq1, load_b_bf16(krow, 32, hi), sc);
#pragma unroll
        for (int r = 0; r < 8; ++r) {
          const int Mv = r + (hi ? 8 : 0);
          const int qi = qt * 16 + Mv;
          const int kj = k0 + nl;
          const float p = (kj > qi) ? 0.0f
                        : __expf(sc[r] * scale - s_m[Mv]) * s_invl[Mv];
          s_pt[wave][Mv * 32 + half * 16 + nl] = p;
        }
      }
    }
    __syncthreads();  // C-layout -> A-layout transpose visibility
    if (act) {
      // async store attn tile straight from LDS (16 rows x 32 f32 = 2KB)
#pragma unroll
      for (int blk = 0; blk < 4; ++blk) {
        const int off = blk * 512 + lane * 16;  // bytes within tile
        const int row = off >> 7;               // /128B per row
        const int col = (off & 127) >> 2;       // float column 0..31
        async_store_b128(attn_row0 + (size_t)row * SEQ + kc * 32 + col,
                         pt_lds + off);
      }
      const float* prow = &s_pt[wave][nl * 32];
      const v16bf ap = load_a_f32(prow, 0, hi);
      const int kg0 = kc * 32;
      __builtin_prefetch(vbase + (size_t)nl * SEQ + kg0 + 128, 0, 3);
      oc0 = wmma_bf16(ap, load_b_bf16(vbase + (size_t)( 0 + nl) * SEQ, kg0, hi), oc0);
      oc1 = wmma_bf16(ap, load_b_bf16(vbase + (size_t)(16 + nl) * SEQ, kg0, hi), oc1);
      oc2 = wmma_bf16(ap, load_b_bf16(vbase + (size_t)(32 + nl) * SEQ, kg0, hi), oc2);
      oc3 = wmma_bf16(ap, load_b_bf16(vbase + (size_t)(48 + nl) * SEQ, kg0, hi), oc3);
    }
    __syncthreads();
  }
  wait_async();  // retire outstanding attn stores before wave exits work

  // merge per-wave partial outputs (ds_add_f32)
#pragma unroll
  for (int r = 0; r < 8; ++r) {
    const int Mv = r + (hi ? 8 : 0);
    atomicAdd(&s_acc[Mv * 64 +  0 + nl], oc0[r]);
    atomicAdd(&s_acc[Mv * 64 + 16 + nl], oc1[r]);
    atomicAdd(&s_acc[Mv * 64 + 32 + nl], oc2[r]);
    atomicAdd(&s_acc[Mv * 64 + 48 + nl], oc3[r]);
  }
  __syncthreads();

  for (int i = tid; i < 16 * 64; i += 128) {
    const int Mv = i >> 6, col = i & 63;
    const int ss = qt * 16 + Mv;
    ctx[((size_t)b * SEQ + ss) * HID + h * 64 + col] = (__bf16)s_acc[i];
  }
}

// ============================================================================
// Kernel 3: out = ctx @ wo^T  (bf16 WMMA, f32 out). One 16x16 tile per wave.
// ============================================================================
__global__ __launch_bounds__(128) void out_proj_kernel(
    const __bf16* __restrict__ ctx, const float* __restrict__ wo,
    float* __restrict__ outp) {
  const int wave = threadIdx.x >> 5;
  const int lane = threadIdx.x & 31;
  const bool hi  = lane >= 16;
  const int tile = blockIdx.x * 4 + wave;
  const int ct = tile % (HID / 16);
  const int rt = tile / (HID / 16);
  const int nl = lane & 15;

  const __bf16* xrow = ctx + (size_t)(rt * 16 + nl) * HID;
  const int j = ct * 16 + nl;
  const float* wrow = wo + (size_t)j * HID;

  v8f c = {};
#pragma unroll 4
  for (int kk = 0; kk < HID; kk += 32) {
    __builtin_prefetch(wrow + kk + 128, 0, 3);
    c = wmma_bf16(load_a_bf16(xrow, kk, hi), load_b_f32(wrow, kk, hi), c);
  }

#pragma unroll
  for (int r = 0; r < 8; ++r)
    outp[(size_t)(rt * 16 + r + (hi ? 8 : 0)) * HID + j] = c[r];
}

// ============================================================================
extern "C" void kernel_launch(void* const* d_in, const int* in_sizes, int n_in,
                              void* d_out, int out_size, void* d_ws, size_t ws_size,
                              hipStream_t stream) {
  const float* X    = (const float*)d_in[0];
  const float* fcos = (const float*)d_in[1];
  const float* fsin = (const float*)d_in[2];
  const float* wq   = (const float*)d_in[3];
  const float* wk   = (const float*)d_in[4];
  const float* wv   = (const float*)d_in[5];
  const float* wo   = (const float*)d_in[6];

  // workspace: qb 12MB | kb 4MB | vbT 4MB | ctx 12MB  (32 MiB total, bf16)
  const size_t QB_ELEMS  = (size_t)BSZ * NH  * SEQ * HD;  // 6,291,456
  const size_t KB_ELEMS  = (size_t)BSZ * NKV * SEQ * HD;  // 2,097,152
  __bf16* qbuf = (__bf16*)d_ws;
  __bf16* kbuf = qbuf + QB_ELEMS;
  __bf16* vbufT = kbuf + KB_ELEMS;
  __bf16* ctx  = vbufT + KB_ELEMS;

  float* outp = (float*)d_out;                    // [B,S,768] f32
  float* attn = outp + (size_t)BS * HID;          // [B,12,S,S] f32

  // 512 row tiles x 80 col tiles, 4 tiles/block
  qkv_rope_kernel<<<dim3((BS / 16) * (NCOL / 16) / 4), dim3(128), 0, stream>>>(
      X, wq, wk, wv, fcos, fsin, qbuf, kbuf, vbufT);

  // one block per (b, h, 16-row q tile): 4*12*128 = 6144
  attn_kernel<<<dim3(BSZ * NH * (SEQ / 16)), dim3(128), 0, stream>>>(
      qbuf, kbuf, vbufT, attn, ctx);

  // 512 x 48 tiles, 4 tiles/block
  out_proj_kernel<<<dim3((BS / 16) * (HID / 16) / 4), dim3(128), 0, stream>>>(
      ctx, wo, outp);
}
